// EnhancedGCN_14181982011663
// MI455X (gfx1250) — matrix operand order repaired and checked by reading, not compiled, not run on hardware
//
#include <hip/hip_runtime.h>
#include <hip/hip_bf16.h>
#include <math.h>

// ---------------------------------------------------------------------------
// EnhancedGCN on MI455X (gfx1250, wave32)
//   - scatter-add over edges with native global_atomic_add_f32 (mem-bound part)
//   - node GEMMs via v_wmma_f32_16x16x32_bf16, A staged in LDS (padded)
//   - fused bias + exact GELU + norm_src scaling in GEMM epilogue
//   - final LayerNorm with wave32 shuffle reductions
// ---------------------------------------------------------------------------

#define NNODES 100000
#define NEDGES 1600000
#define DIM    128
#define LDA    136   // LDS row stride in bf16 elems (272 B) -> conflict-free

typedef __attribute__((ext_vector_type(16))) __bf16 v16bf;
typedef __attribute__((ext_vector_type(8)))  __bf16 v8bf;
typedef __attribute__((ext_vector_type(4)))  __bf16 v4bf;
typedef __attribute__((ext_vector_type(8)))  float  v8f;

// ---------------- utility kernels ----------------

__global__ void zero_f32(float* __restrict__ p, int n4) {
    int i = blockIdx.x * blockDim.x + threadIdx.x;
    if (i < n4) {
        float4 z = {0.f, 0.f, 0.f, 0.f};
        *(float4*)(p + (size_t)i * 4) = z;
    }
}

__global__ void deg_count(const int* __restrict__ idx, float* __restrict__ deg, int E) {
    int i = blockIdx.x * blockDim.x + threadIdx.x;
    if (i < E) unsafeAtomicAdd(&deg[idx[i]], 1.0f);
}

__global__ void deg2norm(float* __restrict__ d, int n) {
    int i = blockIdx.x * blockDim.x + threadIdx.x;
    if (i < n) d[i] = rsqrtf(fmaxf(d[i], 1.0f));
}

__global__ void cvt_bf16(const float* __restrict__ w, __bf16* __restrict__ o, int n) {
    int i = blockIdx.x * blockDim.x + threadIdx.x;
    if (i < n) o[i] = (__bf16)w[i];
}

// hs[row] = x[row] * s[row]  (row-broadcast scale, float4 vectorized)
__global__ void scale_rows(const float* __restrict__ x, const float* __restrict__ s,
                           float* __restrict__ out, int M) {
    int i = blockIdx.x * blockDim.x + threadIdx.x;   // over M*32 float4 quads
    if (i >= M * 32) return;
    int row = i >> 5, c4 = i & 31;
    float sc = s[row];
    float4 v = *(const float4*)(x + (size_t)row * DIM + c4 * 4);
    v.x *= sc; v.y *= sc; v.z *= sc; v.w *= sc;
    *(float4*)(out + (size_t)row * DIM + c4 * 4) = v;
}

// one wave per edge: lane handles 4 contiguous features (float4 gather, 4 f32 atomics)
__global__ __launch_bounds__(256)
void edge_scatter(const int* __restrict__ src, const int* __restrict__ dst,
                  const float* __restrict__ hs, float* __restrict__ agg, int E) {
    long long t = (long long)blockIdx.x * blockDim.x + threadIdx.x;
    int e = (int)(t >> 5);
    if (e >= E) return;
    int lane = (int)(t & 31);
    int s = src[e], d = dst[e];
    const float4 v = *(const float4*)(hs + (size_t)s * DIM + lane * 4);
    float* a = agg + (size_t)d * DIM + lane * 4;
    unsafeAtomicAdd(a + 0, v.x);
    unsafeAtomicAdd(a + 1, v.y);
    unsafeAtomicAdd(a + 2, v.z);
    unsafeAtomicAdd(a + 3, v.w);
}

// ---------------- WMMA GEMM: out = act((ndst*A) @ W + b) [* nsrc] ----------------
// Grid: M/16 blocks of 256 threads (8 waves). Wave w owns output cols [16w,16w+16).
__global__ __launch_bounds__(256)
void gemm_wmma(const float* __restrict__ A, const float* __restrict__ ndst,
               const __bf16* __restrict__ W, const float* __restrict__ bias,
               const float* __restrict__ nsrc, float* __restrict__ out,
               int M, int apply_gelu) {
    __shared__ __bf16 As[16 * LDA];
    const int t  = threadIdx.x;
    const int rb = blockIdx.x * 16;

    // stage 16x128 fp32 tile -> scale by norm_dst -> bf16 in LDS
    for (int i = t; i < 512; i += 256) {          // 16 rows x 32 float4 cols
        int row = i >> 5, c4 = i & 31;
        int gr = rb + row;
        float s = ndst[gr];
        const float4 v = *(const float4*)(A + (size_t)gr * DIM + c4 * 4);
        v4bf b4;
        b4[0] = (__bf16)(v.x * s); b4[1] = (__bf16)(v.y * s);
        b4[2] = (__bf16)(v.z * s); b4[3] = (__bf16)(v.w * s);
        *(v4bf*)(&As[row * LDA + c4 * 4]) = b4;
    }
    __syncthreads();

    const int lane = t & 31;
    const int nb   = (t >> 5) * 16;   // wave's N base
    const int lo   = lane & 15, hi = lane >> 4;

    v8f acc = {};
#pragma unroll
    for (int kk = 0; kk < 4; ++kk) {
        // A fragment (ISA 16-bit A layout): row=lo, K chunks [8hi,8hi+8) and [16+8hi,...)
        const __bf16* arow = &As[lo * LDA + kk * 32 + hi * 8];
        v8bf a0 = *(const v8bf*)(arow);
        v8bf a1 = *(const v8bf*)(arow + 16);
        v16bf a;
#pragma unroll
        for (int i = 0; i < 8; ++i) { a[i] = a0[i]; a[i + 8] = a1[i]; }
        // B fragment: col N = nb+lo, K chunks [8hi,8hi+8) and [16+8hi,...) (matches A)
        v16bf b;
#pragma unroll
        for (int i = 0; i < 8; ++i) {
            b[i]     = W[(size_t)(kk * 32 + hi * 8 + i) * DIM + nb + lo];
            b[i + 8] = W[(size_t)(kk * 32 + 16 + hi * 8 + i) * DIM + nb + lo];
        }
        acc = __builtin_amdgcn_wmma_f32_16x16x32_bf16(false, a, false, b,
                                                      (short)0, acc, false, false);
    }

    // C/D layout: col = nb + lo, row = rb + r + 8*hi
    const int col = nb + lo;
    const float bc = bias[col];
#pragma unroll
    for (int r = 0; r < 8; ++r) {
        int gr = rb + r + hi * 8;
        float v = acc[r] + bc;
        if (apply_gelu) v = 0.5f * v * (1.0f + erff(v * 0.7071067811865475f));
        if (nsrc) v *= nsrc[gr];
        out[(size_t)gr * DIM + col] = v;
    }
}

// ---------------- LayerNorm (in-place), one wave per row ----------------
__global__ __launch_bounds__(256)
void layernorm(float* __restrict__ h, const float* __restrict__ gamma,
               const float* __restrict__ beta, int M) {
    int t = threadIdx.x;
    int lane = t & 31;
    int row = blockIdx.x * 8 + (t >> 5);
    if (row >= M) return;
    float4 v = *(const float4*)(h + (size_t)row * DIM + lane * 4);
    float s = v.x + v.y + v.z + v.w;
    for (int o = 16; o; o >>= 1) s += __shfl_xor(s, o, 32);
    float mu = s * (1.0f / 128.0f);
    float dx = v.x - mu, dy = v.y - mu, dz = v.z - mu, dw = v.w - mu;
    float q = dx * dx + dy * dy + dz * dz + dw * dw;
    for (int o = 16; o; o >>= 1) q += __shfl_xor(q, o, 32);
    float rs = rsqrtf(q * (1.0f / 128.0f) + 1e-5f);
    float4 g  = *(const float4*)(gamma + lane * 4);
    float4 be = *(const float4*)(beta  + lane * 4);
    float4 r;
    r.x = dx * rs * g.x + be.x;
    r.y = dy * rs * g.y + be.y;
    r.z = dz * rs * g.z + be.z;
    r.w = dw * rs * g.w + be.w;
    *(float4*)(h + (size_t)row * DIM + lane * 4) = r;
}

// ---------------- launcher ----------------
extern "C" void kernel_launch(void* const* d_in, const int* in_sizes, int n_in,
                              void* d_out, int out_size, void* d_ws, size_t ws_size,
                              hipStream_t stream) {
    (void)in_sizes; (void)n_in; (void)out_size; (void)ws_size;
    const float* x   = (const float*)d_in[0];
    const int*   src = (const int*)d_in[1];
    const int*   dst = (const int*)d_in[2];
    const float* W[4]  = {(const float*)d_in[3], (const float*)d_in[5],
                          (const float*)d_in[7], (const float*)d_in[9]};
    const float* bv[4] = {(const float*)d_in[4], (const float*)d_in[6],
                          (const float*)d_in[8], (const float*)d_in[10]};
    const float* gamma = (const float*)d_in[11];
    const float* beta  = (const float*)d_in[12];

    char* ws = (char*)d_ws;
    size_t off = 0;
    auto alloc = [&](size_t bytes) -> void* {
        void* p = ws + off;
        off = (off + bytes + 255) & ~(size_t)255;
        return p;
    };
    float*  nsrc = (float*)alloc((size_t)NNODES * 4);
    float*  ndst = (float*)alloc((size_t)NNODES * 4);
    __bf16* Wb   = (__bf16*)alloc((size_t)4 * DIM * DIM * 2);
    float*  hs   = (float*)alloc((size_t)NNODES * DIM * 4);
    float*  agg  = (float*)alloc((size_t)NNODES * DIM * 4);

    const int B = 256;
    // degrees -> inverse-sqrt norms
    zero_f32<<<(NNODES / 4 + B - 1) / B, B, 0, stream>>>(nsrc, NNODES / 4);
    zero_f32<<<(NNODES / 4 + B - 1) / B, B, 0, stream>>>(ndst, NNODES / 4);
    deg_count<<<(NEDGES + B - 1) / B, B, 0, stream>>>(src, nsrc, NEDGES);
    deg_count<<<(NEDGES + B - 1) / B, B, 0, stream>>>(dst, ndst, NEDGES);
    deg2norm<<<(NNODES + B - 1) / B, B, 0, stream>>>(nsrc, NNODES);
    deg2norm<<<(NNODES + B - 1) / B, B, 0, stream>>>(ndst, NNODES);
    // weights -> bf16 once per call
    for (int l = 0; l < 4; ++l)
        cvt_bf16<<<(DIM * DIM + B - 1) / B, B, 0, stream>>>(W[l], Wb + (size_t)l * DIM * DIM, DIM * DIM);
    // hs0 = x * norm_src
    scale_rows<<<(NNODES * 32 + B - 1) / B, B, 0, stream>>>(x, nsrc, hs, NNODES);

    const int n4agg = NNODES * DIM / 4;
    for (int l = 0; l < 4; ++l) {
        zero_f32<<<(n4agg + B - 1) / B, B, 0, stream>>>(agg, n4agg);
        edge_scatter<<<(int)(((long long)NEDGES * 32 + B - 1) / B), B, 0, stream>>>(
            src, dst, hs, agg, NEDGES);
        bool last = (l == 3);
        gemm_wmma<<<NNODES / 16, B, 0, stream>>>(
            agg, ndst, Wb + (size_t)l * DIM * DIM, bv[l],
            last ? nullptr : nsrc, last ? (float*)d_out : hs,
            NNODES, last ? 0 : 1);
    }
    layernorm<<<(NNODES + 7) / 8, B, 0, stream>>>((float*)d_out, gamma, beta, NNODES);
}
